// TreeLSTM_38603166056850
// MI455X (gfx1250) — compile-verified, matrix operand order checked by reading, not run on hardware
//
#include <hip/hip_runtime.h>

typedef __attribute__((ext_vector_type(16))) __bf16 v16bf;
typedef __attribute__((ext_vector_type(8)))  __bf16 v8bf;
typedef __attribute__((ext_vector_type(8)))  float  v8f;

#define HDIM   128   // hidden size
#define KDIM   256   // 2H (gemm K)
#define GDIM   512   // 4H (gemm N, gate outputs)
#define MROWS  32    // nodes per block (2 M-tiles of 16)
#define MTILES 2
#define NTHR   256   // 8 waves

__device__ __forceinline__ __bf16 f2bf(float x) {
  union { float f; unsigned u; } v; v.f = x;
  unsigned r = v.u + 0x7FFFu + ((v.u >> 16) & 1u);   // round-to-nearest-even
  unsigned short s = (unsigned short)(r >> 16);
  return *reinterpret_cast<__bf16*>(&s);
}

__device__ __forceinline__ float sigm(float x) { return 1.0f / (1.0f + __expf(-x)); }

__global__ void convert_w_bf16(const float* __restrict__ W, __bf16* __restrict__ Wb, int n) {
  int i = blockIdx.x * blockDim.x + threadIdx.x;
  if (i < n) Wb[i] = f2bf(W[i]);
}

// One LSTM-cell level.
// mode 0 (leaves): gate input = [0(128) | leaf[node](128)], cc_sum = 0, src = leaf_features (N x 128)
// mode 1 (reduce): s = h[2m]+h[2m+1]; gate input = [s | 0.5*s]; cc_sum = c[2m]+c[2m+1]
__global__ __launch_bounds__(NTHR)
void treelstm_cell(const float* __restrict__ src, const float* __restrict__ cprev,
                   const __bf16* __restrict__ Wb,   // (512 x 256) bf16 row-major (N x K)
                   const float*  __restrict__ bias, // (512)
                   float* __restrict__ hout, float* __restrict__ cout,
                   int N, int mode)
{
  __shared__ __bf16 Ash[MROWS * KDIM];   // 16 KB: bf16 activation tile (M x K)
  __shared__ float  Csh[MROWS * HDIM];   // 16 KB: fp32 cc_sum tile

  const int tid  = threadIdx.x;
  const int row0 = blockIdx.x * MROWS;

  // ---------------- Stage 1: build A (bf16) and cc_sum (f32) in LDS ----------------
  #pragma unroll
  for (int e = 0; e < (MROWS * KDIM) / NTHR; ++e) {
    int idx  = e * NTHR + tid;
    int r    = idx >> 8;              // / KDIM
    int col  = idx & (KDIM - 1);
    int node = row0 + r;
    float a = 0.0f;
    if (node < N) {
      if (mode == 0) {
        if (col >= HDIM) a = src[(size_t)node * HDIM + (col - HDIM)];
      } else {
        int hc = col & (HDIM - 1);
        float s = src[(size_t)(2 * node) * HDIM + hc] +
                  src[(size_t)(2 * node + 1) * HDIM + hc];
        a = (col < HDIM) ? s : 0.5f * s;
      }
    }
    Ash[idx] = f2bf(a);
  }
  #pragma unroll
  for (int e = 0; e < (MROWS * HDIM) / NTHR; ++e) {
    int idx  = e * NTHR + tid;
    int r    = idx >> 7;              // / HDIM
    int hc   = idx & (HDIM - 1);
    int node = row0 + r;
    float cc = 0.0f;
    if (mode == 1 && node < N)
      cc = cprev[(size_t)(2 * node) * HDIM + hc] +
           cprev[(size_t)(2 * node + 1) * HDIM + hc];
    Csh[idx] = cc;
  }
  __syncthreads();

  // ---------------- Stage 2: WMMA gemm ----------------
  // Wave w owns gate-output columns {g*128 + w*16 + (0..15)} for g = f,i,ct,o.
  const int wave = tid >> 5;
  const int lane = tid & 31;
  const int nlo  = lane & 15;          // row (A) / column (B) owned by this lane
  const int koff = (lane >> 4) * 8;    // lanes 16-31 hold K+8 / K+24 chunks

  v8f acc[MTILES][4] = {};

  const __bf16* brow0 = Wb + (size_t)(wave * 16 + nlo) * KDIM + koff;
  #pragma unroll
  for (int kt = 0; kt < KDIM; kt += 32) {
    v16bf afr[MTILES];
    #pragma unroll
    for (int mt = 0; mt < MTILES; ++mt) {
      const __bf16* ap = Ash + (size_t)(mt * 16 + nlo) * KDIM + kt + koff;
      v8bf a0 = *(const v8bf*)(ap);        // K = kt+koff   .. +7   (16B)
      v8bf a1 = *(const v8bf*)(ap + 16);   // K = kt+koff+16.. +23  (16B)
      #pragma unroll
      for (int i = 0; i < 8; ++i) { afr[mt][i] = a0[i]; afr[mt][i + 8] = a1[i]; }
    }
    #pragma unroll
    for (int g = 0; g < 4; ++g) {
      const __bf16* bp = brow0 + (size_t)g * HDIM * KDIM + kt;  // B[k,n] = W[n,k]
      v8bf b0 = *(const v8bf*)(bp);
      v8bf b1 = *(const v8bf*)(bp + 16);
      v16bf bfr;
      #pragma unroll
      for (int i = 0; i < 8; ++i) { bfr[i] = b0[i]; bfr[i + 8] = b1[i]; }
      #pragma unroll
      for (int mt = 0; mt < MTILES; ++mt)
        acc[mt][g] = __builtin_amdgcn_wmma_f32_16x16x32_bf16(
            false, afr[mt], false, bfr, (short)0, acc[mt][g], false, false);
    }
  }

  // ---------------- Stage 3: LSTM nonlinearity + store ----------------
  // C/D layout: VGPR r, lanes 0-15 -> M=r, N=lane; lanes 16-31 -> M=8+r, N=lane-16.
  const int   hc    = wave * 16 + nlo;
  const float bF    = bias[0 * HDIM + hc];
  const float bI    = bias[1 * HDIM + hc];
  const float bC    = bias[2 * HDIM + hc];
  const float bO    = bias[3 * HDIM + hc];
  const int   rbase = (lane >> 4) * 8;

  #pragma unroll
  for (int mt = 0; mt < MTILES; ++mt) {
    #pragma unroll
    for (int r = 0; r < 8; ++r) {
      int m    = mt * 16 + rbase + r;
      int node = row0 + m;
      if (node < N) {
        float fv = sigm(acc[mt][0][r] + bF);
        float iv = sigm(acc[mt][1][r] + bI);
        float gv = tanhf(acc[mt][2][r] + bC);
        float ov = sigm(acc[mt][3][r] + bO);
        float c  = fv * Csh[m * HDIM + hc] + iv * gv;
        float h  = ov * tanhf(c);
        hout[(size_t)node * HDIM + hc] = h;
        cout[(size_t)node * HDIM + hc] = c;
      }
    }
  }
}

extern "C" void kernel_launch(void* const* d_in, const int* in_sizes, int n_in,
                              void* d_out, int out_size, void* d_ws, size_t ws_size,
                              hipStream_t stream) {
  (void)n_in; (void)out_size; (void)ws_size;
  const float* leaf = (const float*)d_in[0];
  const float* W    = (const float*)d_in[1];
  const float* bias = (const float*)d_in[2];
  const int L = in_sizes[0] / HDIM;    // 2^17 in the reference

  char* ws = (char*)d_ws;
  size_t off = 0;
  __bf16* Wb = (__bf16*)(ws + off); off += (size_t)GDIM * KDIM * sizeof(__bf16);
  off = (off + 255) & ~(size_t)255;
  float* hA = (float*)(ws + off); off += (size_t)L * HDIM * sizeof(float);
  float* cA = (float*)(ws + off); off += (size_t)L * HDIM * sizeof(float);
  float* hB = (float*)(ws + off); off += (size_t)(L / 2) * HDIM * sizeof(float);
  float* cB = (float*)(ws + off); off += (size_t)(L / 2) * HDIM * sizeof(float);

  // W -> bf16 once per launch (deterministic)
  convert_w_bf16<<<(GDIM * KDIM + 255) / 256, 256, 0, stream>>>(W, Wb, GDIM * KDIM);

  // Leaf level
  int N = L;
  treelstm_cell<<<(N + MROWS - 1) / MROWS, NTHR, 0, stream>>>(
      leaf, nullptr, Wb, bias, hA, cA, N, 0);

  // 17 reduction levels, ping-pong h/c in workspace; root h -> d_out
  float *hin = hA, *cin = cA, *ho = hB, *co = cB;
  while (N > 1) {
    N >>= 1;
    float* hdst = (N == 1) ? (float*)d_out : ho;
    treelstm_cell<<<(N + MROWS - 1) / MROWS, NTHR, 0, stream>>>(
        hin, cin, Wb, bias, hdst, co, N, 1);
    float* t;
    t = hin; hin = ho; ho = t;
    t = cin; cin = co; co = t;
  }
}
